// Rendering_5884105196257
// MI455X (gfx1250) — matrix operand order repaired and checked by reading, not compiled, not run on hardware
//
#include <hip/hip_runtime.h>

#define HH 512
#define WW 960
#define NN (HH*WW)
#define BB 8
#define CC 3
#define TW 64
#define TH 32

__device__ __constant__ float kBIG = 100000000.0f;
__device__ __constant__ float kTHRESH = 10000.0f;

typedef _Float16 v16h __attribute__((ext_vector_type(16)));
typedef float v8f __attribute__((ext_vector_type(8)));

// ---------------- K0: init workspace ----------------
__global__ void k_init(unsigned* __restrict__ maxu, int* __restrict__ arg,
                       float* __restrict__ m) {
  int i = blockIdx.x * blockDim.x + threadIdx.x;
  if (i < BB * NN) { maxu[i] = 0u; arg[i] = NN; m[i] = 0.0f; }
}

// ---------------- K1: scatter-max of |fx| ----------------
__global__ void k_scatter_max(const float* __restrict__ flow,
                              unsigned* __restrict__ maxu) {
  int i = blockIdx.x * blockDim.x + threadIdx.x;
  if (i >= BB * NN) return;
  int b = i / NN, p = i - b * NN;
  int y = p / WW, x = p - y * WW;
  float fx = flow[i];
  int cx = __float2int_rn((float)x + fx);       // round half-to-even, like jnp.round
  bool inb = (cx >= 0) && (cx < WW);
  int lin = inb ? (y * WW + cx) : 0;            // OOB collapses to lin 0 (matches ref)
  float val = fabsf(fx);                        // >= 0: bits are order-preserving
  atomicMax(&maxu[b * NN + lin], __float_as_uint(val));
}

// ---------------- K2: scatter-min pid among winners ----------------
__global__ void k_scatter_argmin(const float* __restrict__ flow,
                                 const unsigned* __restrict__ maxu,
                                 int* __restrict__ arg) {
  int i = blockIdx.x * blockDim.x + threadIdx.x;
  if (i >= BB * NN) return;
  int b = i / NN, p = i - b * NN;
  int y = p / WW, x = p - y * WW;
  float fx = flow[i];
  int cx = __float2int_rn((float)x + fx);
  bool inb = (cx >= 0) && (cx < WW);
  int lin = inb ? (y * WW + cx) : 0;
  unsigned vb = __float_as_uint(fabsf(fx));
  unsigned g = maxu[b * NN + lin];
  if (vb == g && g != 0u)                       // g!=0u <=> g as float > 0 (nonneg)
    atomicMin(&arg[b * NN + lin], p);
}

// ---------------- K3: gather channels, compose image, mark m ----------------
__global__ void k_gather(const float* __restrict__ x, const int* __restrict__ arg,
                         float* __restrict__ img, float* __restrict__ m) {
  int i = blockIdx.x * blockDim.x + threadIdx.x;
  if (i >= BB * NN) return;
  int b = i / NN, j = i - b * NN;
  int a = arg[i];
  bool valid = a < NN;
  int ac = valid ? a : 0;
  m[b * NN + ac] = 1.0f;                        // duplicate 1.0 stores are benign
#pragma unroll
  for (int ch = 0; ch < CC; ++ch) {
    float v = valid ? x[((size_t)(b * CC + ch)) * NN + ac] : kBIG;
    if (j == 0) v = kBIG;                       // mv[:,0,:] = BIG
    img[((size_t)(b * CC + ch)) * NN + j] = (v < kTHRESH) ? v : 0.0f;
  }
}

// ---------------- K3b: force m at pixel 0 to zero (m[:,0]=0) ----------------
__global__ void k_zero_pix0(float* __restrict__ m) {
  if (threadIdx.x < BB) m[(size_t)threadIdx.x * NN] = 0.0f;
}

// ---------------- K4: 3x3 erosion + edge-replicated crop ----------------
// Async-stage m halo tile into LDS, vertical 3-sums (VALU), horizontal 3-sums
// as a banded-ones GEMM via v_wmma_f32_16x16x32_f16, E = (boxsum == 9).
__global__ void __launch_bounds__(256) k_mask(const float* __restrict__ m,
                                              float* __restrict__ mask) {
  __shared__ float ms[TH + 2][TW + 2];   // 34 x 66 halo tile of m
  __shared__ float vs[TH][TW + 2];       // 32 x 66 vertical 3-sums
  __shared__ float Es[TH][TW];           // 32 x 64 eroded tile

  const int b = blockIdx.z;
  const int r = blockIdx.y * TH;
  const int c = blockIdx.x * TW;
  const int tid = threadIdx.x;
  const float* mb = m + (size_t)b * NN;

  // Stage halo tile: per-lane async copies global -> LDS (ASYNCcnt path).
  for (int idx = tid; idx < (TH + 2) * (TW + 2); idx += 256) {
    int i = idx / (TW + 2), j = idx - i * (TW + 2);
    int gy = min(max(r - 1 + i, 0), HH - 1);   // clamp: values only matter interior
    int gx = min(max(c - 1 + j, 0), WW - 1);
    unsigned long long src = (unsigned long long)(mb + (size_t)gy * WW + gx);
    unsigned dst = (unsigned)(unsigned long long)&ms[i][j];  // LDS byte offset
    asm volatile("global_load_async_to_lds_b32 %0, %1, off"
                 :: "v"(dst), "v"(src) : "memory");
  }
  asm volatile("s_wait_asynccnt 0" ::: "memory");
  __syncthreads();

  // Vertical 3-sums: vs[i][j] = m(r+i-1..r+i+1, c-1+j)
  for (int idx = tid; idx < TH * (TW + 2); idx += 256) {
    int i = idx / (TW + 2), j = idx - i * (TW + 2);
    vs[i][j] = ms[i][j] + ms[i + 1][j] + ms[i + 2][j];
  }
  __syncthreads();

  // Horizontal 3-sum as GEMM: D[mrow][n] = sum_t A[mrow][t] * Band[t][n],
  // A[mrow][t] = vs[sr+mrow][sc+t] (t<18, else 0), Band[t][n] = (t in {n,n+1,n+2}).
  {
    const int wave = tid >> 5, lane = tid & 31;
    const int sr = (wave >> 2) * 16, sc = (wave & 3) * 16;
    const int row = sr + (lane & 15);
    const int n = lane & 15;
    v16h a, bb;
#pragma unroll
    for (int e = 0; e < 16; ++e) {
      // 16-bit A 16x32 layout: lane<16 -> K = {0..7,16..23}; lane>=16 -> {8..15,24..31}
      int tA = (lane < 16) ? ((e < 8) ? e : e + 8) : ((e < 8) ? e + 8 : e + 16);
      a[e] = (_Float16)((tA < 18) ? vs[row][sc + tA] : 0.0f);
      // 16-bit B 32x16 layout: same K order as A (lane<16 -> {0..7,16..23}; lane>=16 -> {8..15,24..31})
      int tB = tA;
      bb[e] = (_Float16)(((tB >= n) && (tB <= n + 2)) ? 1.0f : 0.0f);
    }
    v8f acc = {};
    acc = __builtin_amdgcn_wmma_f32_16x16x32_f16(false, a, false, bb,
                                                 (short)0, acc, false, false);
#pragma unroll
    for (int v = 0; v < 8; ++v) {
      int rl = sr + v + ((lane < 16) ? 0 : 8);   // D layout: VGPR v -> M=v / v+8
      int cl = sc + (lane & 15);
      Es[rl][cl] = (acc[v] > 8.5f) ? 1.0f : 0.0f;  // boxsum==9 -> erosion product
    }
  }
  __syncthreads();

  // mask(h,w) = E(clamp(h,1,H-2), clamp(w,1,W-2)); clamp stays inside this tile.
  float* mo = mask + (size_t)b * NN;
  for (int idx = tid; idx < TH * TW; idx += 256) {
    int i = idx / TW, j = idx - i * TW;
    int h = r + i, w = c + j;
    int hh = min(max(h, 1), HH - 2);
    int ww = min(max(w, 1), WW - 2);
    mo[(size_t)h * WW + w] = Es[hh - r][ww - c];
  }
}

extern "C" void kernel_launch(void* const* d_in, const int* in_sizes, int n_in,
                              void* d_out, int out_size, void* d_ws, size_t ws_size,
                              hipStream_t stream) {
  const float* x = (const float*)d_in[0];      // (8,3,512,960)
  const float* flow = (const float*)d_in[1];   // (8,1,512,960)
  float* img = (float*)d_out;                  // B*C*N floats
  float* mask = img + (size_t)BB * CC * NN;    // B*N floats

  unsigned* maxu = (unsigned*)d_ws;                  // B*N u32
  int* arg = (int*)(maxu + (size_t)BB * NN);         // B*N i32
  float* m = (float*)(arg + (size_t)BB * NN);        // B*N f32

  const int total = BB * NN;
  const int blocks = (total + 255) / 256;
  k_init<<<blocks, 256, 0, stream>>>(maxu, arg, m);
  k_scatter_max<<<blocks, 256, 0, stream>>>(flow, maxu);
  k_scatter_argmin<<<blocks, 256, 0, stream>>>(flow, maxu, arg);
  k_gather<<<blocks, 256, 0, stream>>>(x, arg, img, m);
  k_zero_pix0<<<1, 32, 0, stream>>>(m);
  dim3 g(WW / TW, HH / TH, BB);                // 15 x 16 x 8
  k_mask<<<g, 256, 0, stream>>>(m, mask);
}
